// GLARE_28166395527553
// MI455X (gfx1250) — matrix operand order repaired and checked by reading, not compiled, not run on hardware
//
#include <hip/hip_runtime.h>
#include <hip/hip_bf16.h>
#include <cstdint>

// ---------------------------------------------------------------------------
// MI455X (gfx1250): bf16 WMMA (fp32 acc) for all dense GEMMs, double-buffered
// LDS pipeline, async global->LDS staging (GLOBAL_LOAD_ASYNC_TO_LDS_B128,
// ASYNCcnt-tracked) when the toolchain exposes it.
// fp32 VALU for LN / rope / linear-attention einsums / depthwise conv.
// Workspace requirement: ~260 MiB.
// ---------------------------------------------------------------------------

typedef __bf16 bf16;
typedef __attribute__((ext_vector_type(16))) __bf16 v16bf;
typedef __attribute__((ext_vector_type(8)))  float  v8f;

#define BDIM 1024          // D
#define LDIM 2048          // L
#define BB   4             // B
#define HH   16            // H
#define HD   64            // D/H
#define NTOK (BB*LDIM)     // 8192 tokens

#if __has_builtin(__builtin_amdgcn_global_load_async_to_lds_b128) && \
    __has_builtin(__builtin_amdgcn_s_wait_asynccnt)
#define ASYNC_LDS 1
#else
#define ASYNC_LDS 0
#endif

#if ASYNC_LDS
typedef int v4i __attribute__((vector_size(4 * sizeof(int))));
typedef __attribute__((address_space(1))) v4i* as1_v4i;   // global (__device__)
typedef __attribute__((address_space(3))) v4i* as3_v4i;   // LDS (__shared__)
__device__ __forceinline__ void async_cp16(const void* g, void* lds) {
    // generic->AS1 is bit-identical; generic LDS ptr low 32 bits are the LDS
    // byte offset (ISA 10.2 aperture rules).
    __builtin_amdgcn_global_load_async_to_lds_b128(
        (as1_v4i)(uintptr_t)g,
        (as3_v4i)(uint32_t)(uintptr_t)lds,
        0, 0);
}
#endif

// ---------------- epilogues ----------------
enum { EPI_NONE = 0, EPI_SILU = 1, EPI_GELU = 2, EPI_ELU1 = 3 };

template<int E> __device__ __forceinline__ float apply_epi(float x) {
    if (E == EPI_SILU) return x / (1.f + __expf(-x));
    if (E == EPI_GELU) return 0.5f * x * (1.f + erff(x * 0.70710678118654752f));
    if (E == EPI_ELU1) return (x > 0.f) ? (x + 1.f) : __expf(x);   // elu(x)+1
    return x;
}

__device__ __forceinline__ float silu_f(float x) { return x / (1.f + __expf(-x)); }

// ---------------- WMMA GEMM: C[M,N] = A[M,K] @ Bt[N,K]^T + bias, epilogue ---
// A, Bt bf16 row-major.  Block tile 128x128x32, 256 threads = 8 waves (wave32),
// wave grid 2(M) x 4(N), each wave 64x32 -> 4x2 = 8 v_wmma_f32_16x16x32_bf16.
// Double-buffered LDS, one barrier per K-step, loads for tile kb+1 issued
// before computing tile kb.
#define BM 128
#define BN 128
#define BK 32
#define LDT 40   // LDS row stride in bf16 elems (32 + 8 pad, keeps 16B align)

__device__ __forceinline__ v16bf frag_a(const bf16* s, int row0, int lane) {
    // 16-bit A 16x32: lane holds row M=lane&15; half=lane>>4 selects
    // K chunks [half*8, half*8+8) and [16+half*8, ...+8)   (ISA 7.12.2)
    const int r = row0 + (lane & 15);
    const int half = lane >> 4;
    union { uint4 q[2]; v16bf v; } u;
    u.q[0] = *(const uint4*)&s[r * LDT + half * 8];
    u.q[1] = *(const uint4*)&s[r * LDT + 16 + half * 8];
    return u.v;
}

__device__ __forceinline__ v16bf frag_b(const bf16* s, int row0, int lane) {
    // 16-bit B 32x16: lane holds col N=lane&15; half=lane>>4 selects
    // K chunks [half*8, half*8+8) and [16+half*8, ...+8)
    const int r = row0 + (lane & 15);
    const int half = lane >> 4;
    union { uint4 q[2]; v16bf v; } u;
    u.q[0] = *(const uint4*)&s[r * LDT + half * 8];
    u.q[1] = *(const uint4*)&s[r * LDT + 16 + half * 8];
    return u.v;
}

template<int EPI, bool WF, bool WB>
__global__ __launch_bounds__(256)
void k_gemm(const bf16* __restrict__ A, const bf16* __restrict__ Bt,
            const float* __restrict__ bias,
            float* __restrict__ Cf, bf16* __restrict__ Cb,
            int N, int K)
{
    __shared__ __align__(16) bf16 As[2][BM * LDT];
    __shared__ __align__(16) bf16 Bs[2][BN * LDT];

    const int t    = threadIdx.x;
    const int lane = t & 31;
    const int wave = t >> 5;
    const int wrow = (wave & 1) * 64;   // wave tile: 64 rows
    const int wcol = (wave >> 1) * 32;  //            32 cols
    const int m0 = blockIdx.y * BM;
    const int n0 = blockIdx.x * BN;

    // Staging: thread t owns 16B chunks at rows r0 and r0+64, col chunk c0.
    const int r0 = t >> 2;
    const int c0 = (t & 3) * 8;
    const bf16* gA = A  + (size_t)(m0 + r0) * K + c0;
    const bf16* gB = Bt + (size_t)(n0 + r0) * K + c0;
    const size_t rskip = (size_t)64 * K;
    bf16* sA = &As[0][r0 * LDT + c0];
    bf16* sB = &Bs[0][r0 * LDT + c0];

    const v8f vzero = {0.f,0.f,0.f,0.f,0.f,0.f,0.f,0.f};
    v8f acc[4][2];
    #pragma unroll
    for (int i = 0; i < 4; ++i) { acc[i][0] = vzero; acc[i][1] = vzero; }

    const int nk = K / BK;
    int buf = 0;

#if ASYNC_LDS
    // ---- async global->LDS pipeline (ASYNCcnt-tracked, no VGPR round trip)
    async_cp16(gA,         sA);
    async_cp16(gA + rskip, sA + 64 * LDT);
    async_cp16(gB,         sB);
    async_cp16(gB + rskip, sB + 64 * LDT);
    for (int kb = 0; kb < nk; ++kb) {
        __builtin_amdgcn_s_wait_asynccnt(0);
        __syncthreads();
        if (kb + 1 < nk) {
            const int ko = (kb + 1) * BK;
            const int so = (buf ^ 1) * (BM * LDT);
            async_cp16(gA + ko,         sA + so);
            async_cp16(gA + ko + rskip, sA + so + 64 * LDT);
            async_cp16(gB + ko,         sB + so);
            async_cp16(gB + ko + rskip, sB + so + 64 * LDT);
        }
        const bf16* ap = &As[buf][0];
        const bf16* bp = &Bs[buf][0];
        v16bf bf0 = frag_b(bp, wcol,      lane);
        v16bf bf1 = frag_b(bp, wcol + 16, lane);
        #pragma unroll
        for (int i = 0; i < 4; ++i) {
            v16bf af = frag_a(ap, wrow + i * 16, lane);
            acc[i][0] = __builtin_amdgcn_wmma_f32_16x16x32_bf16(
                false, af, false, bf0, (short)0, acc[i][0], false, false);
            acc[i][1] = __builtin_amdgcn_wmma_f32_16x16x32_bf16(
                false, af, false, bf1, (short)0, acc[i][1], false, false);
        }
        buf ^= 1;
    }
#else
    // ---- fallback: reg-staged double buffering (unconditional 16B chunks)
    uint4 ra0, ra1, rb0, rb1;
    ra0 = *(const uint4*)(gA);
    ra1 = *(const uint4*)(gA + rskip);
    rb0 = *(const uint4*)(gB);
    rb1 = *(const uint4*)(gB + rskip);
    for (int kb = 0; kb < nk; ++kb) {
        const int so = buf * (BM * LDT);
        *(uint4*)(sA + so)            = ra0;
        *(uint4*)(sA + so + 64 * LDT) = ra1;
        *(uint4*)(sB + so)            = rb0;
        *(uint4*)(sB + so + 64 * LDT) = rb1;
        __syncthreads();
        if (kb + 1 < nk) {
            const int ko = (kb + 1) * BK;
            ra0 = *(const uint4*)(gA + ko);
            ra1 = *(const uint4*)(gA + ko + rskip);
            rb0 = *(const uint4*)(gB + ko);
            rb1 = *(const uint4*)(gB + ko + rskip);
        }
        const bf16* ap = &As[buf][0];
        const bf16* bp = &Bs[buf][0];
        v16bf bf0 = frag_b(bp, wcol,      lane);
        v16bf bf1 = frag_b(bp, wcol + 16, lane);
        #pragma unroll
        for (int i = 0; i < 4; ++i) {
            v16bf af = frag_a(ap, wrow + i * 16, lane);
            acc[i][0] = __builtin_amdgcn_wmma_f32_16x16x32_bf16(
                false, af, false, bf0, (short)0, acc[i][0], false, false);
            acc[i][1] = __builtin_amdgcn_wmma_f32_16x16x32_bf16(
                false, af, false, bf1, (short)0, acc[i][1], false, false);
        }
        buf ^= 1;
    }
#endif

    // Epilogue: C layout — vgpr r, lane: row = r + 8*(lane>>4), col = lane&15
    const int half = lane >> 4, nn = lane & 15;
    #pragma unroll
    for (int i = 0; i < 4; ++i) {
        #pragma unroll
        for (int j = 0; j < 2; ++j) {
            const int col = n0 + wcol + j * 16 + nn;
            const float bi = bias[col];
            #pragma unroll
            for (int r = 0; r < 8; ++r) {
                const int row = m0 + wrow + i * 16 + half * 8 + r;
                const float v = apply_epi<EPI>(acc[i][j][r] + bi);
                if (WF) Cf[(size_t)row * N + col] = v;
                if (WB) Cb[(size_t)row * N + col] = (bf16)v;
            }
        }
    }
}

// ---------------- transpose + fp32->bf16 cast of weights: Wt[n,k]=W[k,n] ----
__global__ __launch_bounds__(256)
void k_trcast(const float* __restrict__ W, bf16* __restrict__ Wt, int K, int N)
{
    __shared__ float tile[32][33];
    const int n0 = blockIdx.x * 32, k0 = blockIdx.y * 32;
    const int tx = threadIdx.x & 31, ty = threadIdx.x >> 5;
    for (int r = ty; r < 32; r += 8)
        tile[r][tx] = W[(size_t)(k0 + r) * N + n0 + tx];
    __syncthreads();
    for (int r = ty; r < 32; r += 8)
        Wt[(size_t)(n0 + r) * K + k0 + tx] = (bf16)tile[tx][r];
}

// ---------------- layernorm over D=1024, one row per block, bf16 out -------
__global__ __launch_bounds__(256)
void k_layernorm(const float* __restrict__ x, const float* __restrict__ g,
                 const float* __restrict__ b, bf16* __restrict__ out)
{
    const int row = blockIdx.x;
    const int t = threadIdx.x;
    __shared__ float red[8];
    __shared__ float mu_s, rs_s;

    const float4 v = ((const float4*)(x + (size_t)row * BDIM))[t];
    float s = v.x + v.y + v.z + v.w;
    for (int o = 16; o > 0; o >>= 1) s += __shfl_down(s, o);
    if ((t & 31) == 0) red[t >> 5] = s;
    __syncthreads();
    if (t == 0) { float tot = 0; for (int i = 0; i < 8; ++i) tot += red[i];
                  mu_s = tot / (float)BDIM; }
    __syncthreads();
    const float mu = mu_s;
    const float dx = v.x - mu, dy = v.y - mu, dz = v.z - mu, dw = v.w - mu;
    float sq = dx*dx + dy*dy + dz*dz + dw*dw;
    __syncthreads();
    for (int o = 16; o > 0; o >>= 1) sq += __shfl_down(sq, o);
    if ((t & 31) == 0) red[t >> 5] = sq;
    __syncthreads();
    if (t == 0) { float tot = 0; for (int i = 0; i < 8; ++i) tot += red[i];
                  rs_s = rsqrtf(tot / (float)BDIM + 1e-12f); }
    __syncthreads();
    const float rs = rs_s;
    const float4 gv = ((const float4*)g)[t];
    const float4 bv = ((const float4*)b)[t];
    bf16* o4 = out + (size_t)row * BDIM + t * 4;
    o4[0] = (bf16)(dx * rs * gv.x + bv.x);
    o4[1] = (bf16)(dy * rs * gv.y + bv.y);
    o4[2] = (bf16)(dz * rs * gv.z + bv.z);
    o4[3] = (bf16)(dw * rs * gv.w + bv.w);
}

// ---------------- rope tables: cos/sin[l, i] for i<512 ----------------------
__global__ __launch_bounds__(256)
void k_rope_tables(float* __restrict__ cosT, float* __restrict__ sinT)
{
    const int l = blockIdx.x;
    for (int i = threadIdx.x; i < 512; i += 256) {
        const float theta = __expf(-(float)i * (9.210340371976184f / 512.f)); // 10000^{-i/512}
        const float ang = (float)l * theta;
        cosT[l * 512 + i] = cosf(ang);
        sinT[l * 512 + i] = sinf(ang);
    }
}

// ---------------- rope in place on qk (q cols [0,1024), k cols [1024,2048)) -
__global__ __launch_bounds__(256)
void k_rope(float* __restrict__ qk, const float* __restrict__ cosT,
            const float* __restrict__ sinT)
{
    const int idx = blockIdx.x * 256 + threadIdx.x;   // pair index, NTOK*1024
    const int n = idx >> 10, j = idx & 1023;
    const int l = n & (LDIM - 1);
    const int i = j & 511;
    const int col = (j < 512) ? (2 * j) : (1024 + 2 * (j - 512));
    const float c = cosT[l * 512 + i], s = sinT[l * 512 + i];
    float* p = qk + (size_t)n * 2048 + col;
    const float xr = p[0], xi = p[1];
    p[0] = xr * c - xi * s;
    p[1] = xr * s + xi * c;
}

// ---------------- k_mean[b,h,d] = mean_l k[b,l,h,d] (pre-rope) --------------
__global__ __launch_bounds__(256)
void k_kmean(const float* __restrict__ qk, float* __restrict__ kmean)
{
    const int bh = blockIdx.x;             // b*16+h
    const int b = bh >> 4, h = bh & 15;
    const int d = threadIdx.x & 63, slice = threadIdx.x >> 6;
    __shared__ float red[256];
    const float* kp = qk + (size_t)b * LDIM * 2048 + 1024 + h * HD + d;
    float s = 0.f;
    for (int l = slice * 512; l < (slice + 1) * 512; ++l)
        s += kp[(size_t)l * 2048];
    red[threadIdx.x] = s;
    __syncthreads();
    if (slice == 0)
        kmean[bh * HD + d] = (red[d] + red[64 + d] + red[128 + d] + red[192 + d])
                             * (1.f / (float)LDIM);
}

// ---------------- z[b,h,l] = 1/(q.kmean + eps)  (pre-rope q) ----------------
__global__ __launch_bounds__(256)
void k_z(const float* __restrict__ qk, const float* __restrict__ kmean,
         float* __restrict__ z)
{
    const int idx = blockIdx.x * 256 + threadIdx.x;   // B*H*L
    const int l = idx & (LDIM - 1), bh = idx >> 11;
    const int b = bh >> 4, h = bh & 15;
    const float* qp = qk + (size_t)(b * LDIM + l) * 2048 + h * HD;
    const float* km = kmean + bh * HD;
    float s = 0.f;
    #pragma unroll 8
    for (int d = 0; d < HD; ++d) s += qp[d] * km[d];
    z[bh * LDIM + l] = 1.f / (s + 1e-6f);
}

// ---------------- kv[b,h,d,e] = sum_l k_rope[l,d]*v[l,e] / L ---------------
__global__ __launch_bounds__(256)
void k_kv(const float* __restrict__ qk, const float* __restrict__ v,
          float* __restrict__ kv)
{
    const int bh = blockIdx.x;
    const int b = bh >> 4, h = bh & 15;
    const int t = threadIdx.x;
    const int d = t & 63, e0 = (t >> 6) * 16;
    __shared__ float Ks[16][64], Vs[16][64];
    float acc[16];
    #pragma unroll
    for (int j = 0; j < 16; ++j) acc[j] = 0.f;

    for (int l0 = 0; l0 < LDIM; l0 += 16) {
        for (int i = t; i < 16 * 64; i += 256) {
            const int li = i >> 6, dd = i & 63;
            Ks[li][dd] = qk[(size_t)(b * LDIM + l0 + li) * 2048 + 1024 + h * HD + dd];
            Vs[li][dd] = v [(size_t)(b * LDIM + l0 + li) * 1024 + h * HD + dd];
        }
        __syncthreads();
        #pragma unroll
        for (int li = 0; li < 16; ++li) {
            const float kd = Ks[li][d];
            #pragma unroll
            for (int j = 0; j < 16; ++j) acc[j] += kd * Vs[li][e0 + j];
        }
        __syncthreads();
    }
    #pragma unroll
    for (int j = 0; j < 16; ++j)
        kv[((size_t)bh * HD + d) * HD + e0 + j] = acc[j] * (1.f / (float)LDIM);
}

// ---------------- causal depthwise conv (kernel 4) + SiLU -> attn -----------
__global__ __launch_bounds__(256)
void k_conv(const float* __restrict__ v, const float* __restrict__ cw,
            const float* __restrict__ cb, float* __restrict__ attn)
{
    const int idx = blockIdx.x * 256 + threadIdx.x;   // B*L*D
    const int c = idx & 1023, l = (idx >> 10) & (LDIM - 1);
    float acc = cb[c];
    #pragma unroll
    for (int j = 0; j < 4; ++j) {
        const int ll = l - 3 + j;
        if (ll >= 0) acc += v[idx + (j - 3) * 1024] * cw[c * 4 + j];
    }
    attn[idx] = silu_f(acc);
}

// ---------------- attn += z * (q_rope @ kv) ---------------------------------
__global__ __launch_bounds__(256)
void k_out_attn(const float* __restrict__ qk, const float* __restrict__ kv,
                const float* __restrict__ z, float* __restrict__ attn)
{
    const int n = blockIdx.x;            // token (b*L + l)
    const int b = n >> 11, l = n & (LDIM - 1);
    const int t = threadIdx.x;
    __shared__ float qs[1024];
    for (int i = t; i < 1024; i += 256) qs[i] = qk[(size_t)n * 2048 + i];
    __syncthreads();
    for (int o = t; o < 1024; o += 256) {
        const int h = o >> 6, e = o & 63;
        const float* kvp = kv + (size_t)(b * HH + h) * HD * HD + e;
        float s = 0.f;
        #pragma unroll 8
        for (int d = 0; d < HD; ++d) s += qs[h * HD + d] * kvp[d * HD];
        attn[(size_t)n * 1024 + o] += s * z[(b * HH + h) * LDIM + l];
    }
}

// ---------------- misc elementwise ------------------------------------------
__global__ __launch_bounds__(256)
void k_mul_cast(const float* __restrict__ a, const float* __restrict__ b,
                bf16* __restrict__ o)
{
    const int i = blockIdx.x * 256 + threadIdx.x;
    o[i] = (bf16)(a[i] * b[i]);
}

__global__ __launch_bounds__(256)
void k_add(const float* __restrict__ a, const float* __restrict__ b,
           float* __restrict__ o)
{
    const int i = blockIdx.x * 256 + threadIdx.x;
    o[i] = a[i] + b[i];
}

// ===========================================================================
extern "C" void kernel_launch(void* const* d_in, const int* in_sizes, int n_in,
                              void* d_out, int out_size, void* d_ws, size_t ws_size,
                              hipStream_t stream)
{
    const float* x      = (const float*)d_in[0];
    const float* ln1_g  = (const float*)d_in[1];
    const float* ln1_b  = (const float*)d_in[2];
    const float* ln2_g  = (const float*)d_in[3];
    const float* ln2_b  = (const float*)d_in[4];
    const float* W_act  = (const float*)d_in[5];
    const float* b_act  = (const float*)d_in[6];
    const float* W_in   = (const float*)d_in[7];
    const float* b_in   = (const float*)d_in[8];
    const float* W_qk   = (const float*)d_in[9];
    const float* b_qk   = (const float*)d_in[10];
    const float* conv_w = (const float*)d_in[11];
    const float* conv_b = (const float*)d_in[12];
    const float* W_out  = (const float*)d_in[13];
    const float* b_out  = (const float*)d_in[14];
    const float* W1     = (const float*)d_in[15];
    const float* b1     = (const float*)d_in[16];
    const float* W2     = (const float*)d_in[17];
    const float* b2     = (const float*)d_in[18];
    float* out = (float*)d_out;

    // -------- workspace layout (~260 MiB) --------
    char* base = (char*)d_ws;
    size_t off = 0;
    auto alloc = [&](size_t bytes) -> char* {
        char* p = base + off; off += (bytes + 255) & ~(size_t)255; return p;
    };
    const size_t ND  = (size_t)NTOK * BDIM;
    bf16*  wact_t = (bf16*)alloc(2ull*1024*1024);
    bf16*  win_t  = (bf16*)alloc(2ull*1024*1024);
    bf16*  wqk_t  = (bf16*)alloc(4ull*1024*1024);
    bf16*  wout_t = (bf16*)alloc(2ull*1024*1024);
    bf16*  w1_t   = (bf16*)alloc(8ull*1024*1024);
    bf16*  w2_t   = (bf16*)alloc(8ull*1024*1024);
    float* cosT   = (float*)alloc((size_t)LDIM*512*4);
    float* sinT   = (float*)alloc((size_t)LDIM*512*4);
    bf16*  h_bf   = (bf16*)alloc(ND*2);          // ln1 out; later reused: (attn*act) bf16
    bf16*  hin_bf = (bf16*)alloc(ND*2);          // W_in out bf16; later reused: ln2 out
    float* actres = (float*)alloc(ND*4);         // silu(h@W_act)
    float* hin_f  = (float*)alloc(ND*4);         // h@W_in fp32 (v); later reused: h2, then m2
    float* qk     = (float*)alloc(ND*2*4);       // qk fp32 (64MiB); later reused: mlp hidden bf16
    float* attn   = (float*)alloc(ND*4);         // conv + linear-attn out
    float* x2     = (float*)alloc(ND*4);         // shortcut + h
    float* zbuf   = (float*)alloc((size_t)BB*HH*LDIM*4);
    float* kmean  = (float*)alloc((size_t)BB*HH*HD*4);
    float* kvbuf  = (float*)alloc((size_t)BB*HH*HD*HD*4);
    bf16*  mlp_bf = (bf16*)qk;                   // 8192x4096 bf16 == 64 MiB, reuses qk
    float* h2     = hin_f;
    float* m2     = hin_f;
    bf16*  gin_bf = h_bf;
    bf16*  m_bf   = hin_bf;

    const dim3 blk(256);

    // 0) weights -> bf16 transposed (N x K)
    k_trcast<<<dim3(32, 32),  blk, 0, stream>>>(W_act, wact_t, 1024, 1024);
    k_trcast<<<dim3(32, 32),  blk, 0, stream>>>(W_in,  win_t,  1024, 1024);
    k_trcast<<<dim3(64, 32),  blk, 0, stream>>>(W_qk,  wqk_t,  1024, 2048);
    k_trcast<<<dim3(32, 32),  blk, 0, stream>>>(W_out, wout_t, 1024, 1024);
    k_trcast<<<dim3(128, 32), blk, 0, stream>>>(W1,    w1_t,   1024, 4096);
    k_trcast<<<dim3(32, 128), blk, 0, stream>>>(W2,    w2_t,   4096, 1024);
    k_rope_tables<<<dim3(LDIM), blk, 0, stream>>>(cosT, sinT);

    // 1) h = LN1(x)  (bf16)
    k_layernorm<<<dim3(NTOK), blk, 0, stream>>>(x, ln1_g, ln1_b, h_bf);

    // 2) act_res = silu(h@W_act + b_act)   [WMMA]
    k_gemm<EPI_SILU, true, false><<<dim3(1024/BN, NTOK/BM), blk, 0, stream>>>(
        h_bf, wact_t, b_act, actres, (bf16*)nullptr, 1024, 1024);

    // 3) hin = h@W_in + b_in  (fp32 + bf16)   [WMMA]
    k_gemm<EPI_NONE, true, true><<<dim3(1024/BN, NTOK/BM), blk, 0, stream>>>(
        h_bf, win_t, b_in, hin_f, hin_bf, 1024, 1024);

    // 4) qk = elu(hin@W_qk + b_qk)+1   [WMMA, fused elu+1]
    k_gemm<EPI_ELU1, true, false><<<dim3(2048/BN, NTOK/BM), blk, 0, stream>>>(
        hin_bf, wqk_t, b_qk, qk, (bf16*)nullptr, 2048, 1024);

    // 5) k_mean, z from pre-rope q,k; then rope in place
    k_kmean<<<dim3(BB*HH), blk, 0, stream>>>(qk, kmean);
    k_z<<<dim3(BB*HH*LDIM/256), blk, 0, stream>>>(qk, kmean, zbuf);
    k_rope<<<dim3((unsigned)(ND / 256)), blk, 0, stream>>>(qk, cosT, sinT); // NTOK*1024 pairs
    // 6) kv = (k_rope^T v)/L per (b,h)
    k_kv<<<dim3(BB*HH), blk, 0, stream>>>(qk, hin_f, kvbuf);
    // 7) attn = silu(depthwise_conv(v)); attn += z * (q_rope @ kv)
    k_conv<<<dim3((unsigned)(ND / 256)), blk, 0, stream>>>(hin_f, conv_w, conv_b, attn);
    k_out_attn<<<dim3(NTOK), blk, 0, stream>>>(qk, kvbuf, zbuf, attn);

    // 8) h2 = (attn * act_res)@W_out + b_out   [WMMA]
    k_mul_cast<<<dim3((unsigned)(ND / 256)), blk, 0, stream>>>(attn, actres, gin_bf);
    k_gemm<EPI_NONE, true, false><<<dim3(1024/BN, NTOK/BM), blk, 0, stream>>>(
        gin_bf, wout_t, b_out, h2, (bf16*)nullptr, 1024, 1024);

    // 9) x2 = x + h2 ; m = LN2(x2)
    k_add<<<dim3((unsigned)(ND / 256)), blk, 0, stream>>>(x, h2, x2);
    k_layernorm<<<dim3(NTOK), blk, 0, stream>>>(x2, ln2_g, ln2_b, m_bf);

    // 10) mlp = gelu(m@W1 + b1)  (bf16 out)   [WMMA, fused exact GELU]
    k_gemm<EPI_GELU, false, true><<<dim3(4096/BN, NTOK/BM), blk, 0, stream>>>(
        m_bf, w1_t, b1, (float*)nullptr, mlp_bf, 4096, 1024);

    // 11) m2 = mlp@W2 + b2   [WMMA] ; out = x2 + m2
    k_gemm<EPI_NONE, true, false><<<dim3(1024/BN, NTOK/BM), blk, 0, stream>>>(
        mlp_bf, w2_t, b2, m2, (bf16*)nullptr, 1024, 4096);
    k_add<<<dim3((unsigned)(ND / 256)), blk, 0, stream>>>(x2, m2, out);
}
